// LSTM_43396349559296
// MI455X (gfx1250) — compile-verified, matrix operand order checked by reading, not run on hardware
//
#include <hip/hip_runtime.h>

// ---------------------------------------------------------------------------
// LSTM on MI455X (gfx1250): persistent cooperative kernel, bf16 WMMA GEMMs,
// async global->LDS staging (ASYNCcnt) with double buffering.
//
//   gates_t = [x_t | h_{t-1}] * [Wih | Whh]^T + (bias_ih + bias_hh)
//   c_t = sig(f)*c + sig(i)*tanh(g);  h_t = sig(o)*tanh(c_t)
//
//   * 64 WGs x 256 threads (8 wave32s).  WG owns 16 hidden columns across all
//     4 gates -> cell update is WG-local, ONE grid barrier per timestep.
//   * wave v: gate = v&3, M-tiles {2*(v>>2), 2*(v>>2)+1} -> 2 v8f accums.
//   * inputs converted to bf16 once (init kernel) -> recurrence stages pure
//     bf16 tiles via GLOBAL_LOAD_ASYNC_TO_LDS_B128, double-buffered so the
//     next K-chunk's copy overlaps the current chunk's WMMAs.
//   * B operand (weight rows, L2-resident bf16) loaded straight to VGPRs:
//     each W row feeds two waves (v, v+4) of one WG -> read twice per step.
//   * h double-buffered in ws (ping-pong on t); c lives in registers.
// ---------------------------------------------------------------------------

#define B_   64
#define S_   512
#define D_   1024
#define H_   1024
#define G4_  4096
#define NWG  64
#define TPB  256
#define KC   256      // K-chunk staged in LDS
#define KCP  264      // padded LDS row stride (elements) to spread banks

typedef __bf16 bf16;
typedef __attribute__((ext_vector_type(16))) __bf16 v16bf;
typedef __attribute__((ext_vector_type(8)))  __bf16 v8bf;
typedef __attribute__((ext_vector_type(8)))  float  v8f;

__device__ __forceinline__ v16bf bf_cat(v8bf lo, v8bf hi) {
  return __builtin_shufflevector(lo, hi, 0,1,2,3,4,5,6,7,8,9,10,11,12,13,14,15);
}

__device__ __forceinline__ float sigm(float x) { return 1.0f / (1.0f + __expf(-x)); }

// ---- one-time setup: all fp32 -> bf16 conversions, bias sum, barrier = 0
__global__ __launch_bounds__(256) void lstm_init(
    const float* __restrict__ x,   const float* __restrict__ wih,
    const float* __restrict__ whh, const float* __restrict__ h0,
    const float* __restrict__ bih, const float* __restrict__ bhh,
    bf16* __restrict__ xbf, bf16* __restrict__ wih_bf, bf16* __restrict__ whh_bf,
    bf16* __restrict__ hbuf, float* __restrict__ bsum, unsigned* __restrict__ cnt)
{
  size_t i = (size_t)blockIdx.x * 256 + threadIdx.x;   // grid covers B*S*D
  if (i < (size_t)B_ * S_ * D_) xbf[i] = (bf16)x[i];
  if (i < (size_t)G4_ * D_) { wih_bf[i] = (bf16)wih[i]; whh_bf[i] = (bf16)whh[i]; }
  if (i < (size_t)B_ * H_)  hbuf[i] = (bf16)h0[i];     // ping buffer 0 = h_{-1}
  if (i < G4_)              bsum[i] = bih[i] + bhh[i];
  if (i == 0)               *cnt = 0u;
}

// A operand fetch from LDS: 16-bit A-matrix 16x32 layout (ISA 7.12.2):
// lanes 0-15 hold K {k..k+7, k+16..k+23}, lanes 16-31 hold {k+8..15, k+24..31}
__device__ __forceinline__ v16bf lds_a(const bf16* ab, int m, int ln, int hi, int kl) {
  const bf16* p = ab + (m * 16 + ln) * KCP + kl + hi * 8;
  v8bf lo = *(const v8bf*)p;
  v8bf hh = *(const v8bf*)(p + 16);
  return bf_cat(lo, hh);
}

__device__ __forceinline__ void grid_barrier(unsigned* cnt, unsigned target) {
  __threadfence();          // flush this wave's global writes (h ping-pong)
  __syncthreads();
  if (threadIdx.x == 0) {
    __hip_atomic_fetch_add(cnt, 1u, __ATOMIC_ACQ_REL, __HIP_MEMORY_SCOPE_AGENT);
    while (__hip_atomic_load(cnt, __ATOMIC_ACQUIRE, __HIP_MEMORY_SCOPE_AGENT) < target)
      __builtin_amdgcn_s_sleep(1);
  }
  __syncthreads();
}

__global__ __launch_bounds__(TPB, 1) void lstm_persistent(
    const bf16*  __restrict__ xbf,    // [B,S,D] bf16
    const float* __restrict__ c0,     // [B,H]   fp32
    const bf16*  __restrict__ Wih,    // [4H,D]  bf16 row-major
    const bf16*  __restrict__ Whh,    // [4H,H]  bf16 row-major
    bf16*        __restrict__ hbuf,   // [2][B*H] bf16 ping-pong
    const float* __restrict__ bsum,   // [4H]
    unsigned*    __restrict__ cnt,
    float*       __restrict__ out)    // outputs | hn | cn
{
  __shared__ __align__(16) bf16 abuf[2][B_ * KCP];  // double-buffered A (~66 KB)
  __shared__ float gbuf[4 * B_ * 16];               // gate preacts (16 KB)

  const int tid  = threadIdx.x;
  const int wg   = blockIdx.x;
  const int lane = tid & 31;
  const int wv   = tid >> 5;           // wave 0..7
  const int gate = wv & 3;
  const int m0   = (wv >> 2) * 2;      // M-tiles {m0, m0+1}
  const int jb   = wg * 16;            // hidden-column block
  const int ln   = lane & 15;
  const int hi   = lane >> 4;          // K-chunk select within lane pair

  // B operand: column n of B-tile == row (gate*H + jb + ln) of W, contiguous in K.
  const int wrow = gate * H_ + jb + ln;
  const bf16* pWihRow = Wih + (size_t)wrow * D_ + hi * 8;
  const bf16* pWhhRow = Whh + (size_t)wrow * H_ + hi * 8;
  const float biasv = bsum[wrow];

  // c state in registers: thread owns 4 (b, j) cells for the whole sequence.
  float creg[4];
#pragma unroll
  for (int e = 0; e < 4; ++e) {
    int idx = tid + e * TPB;
    int b = idx >> 4, jj = idx & 15;
    creg[e] = c0[b * H_ + jb + jj];
  }

  const size_t OUTS = (size_t)B_ * S_ * H_;

  for (int t = 0; t < S_; ++t) {
    const bf16* hrd = hbuf + (size_t)(t & 1) * (B_ * H_);

    // async-copy one K-chunk of A ([64 x 256] bf16) into LDS buffer `dst`
    auto stage_async = [&](int chunk, bf16* dst) {
      const int kg = chunk * KC;
      const bf16* src; size_t rstride;
      if (kg < D_) { src = xbf + (size_t)t * D_ + kg; rstride = (size_t)S_ * D_; }
      else         { src = hrd + (kg - D_);           rstride = (size_t)H_; }
#pragma unroll
      for (int g2 = 0; g2 < 8; ++g2) {
        int gi  = tid + g2 * TPB;          // 0..2047 groups of 8 elements
        int row = gi >> 5;
        int c8  = (gi & 31) << 3;
        unsigned ldsoff = (unsigned)(unsigned long long)(dst + row * KCP + c8);
        unsigned long long gaddr =
            (unsigned long long)(src + (size_t)row * rstride + c8);
        // per-lane 16B async copy, tracked by ASYNCcnt (ISA 10.7 / 15.18.3)
        asm volatile("global_load_async_to_lds_b128 %0, %1, off"
                     :: "v"(ldsoff), "v"(gaddr) : "memory");
      }
    };

    v8f acc0, acc1;
#pragma unroll
    for (int r = 0; r < 8; ++r) { acc0[r] = biasv; acc1[r] = biasv; }

    // prologue: stage chunk 0
    stage_async(0, abuf[0]);
    asm volatile("s_wait_asynccnt 0x0" ::: "memory");
    __syncthreads();

#pragma unroll 1
    for (int chunk = 0; chunk < (D_ + H_) / KC; ++chunk) {
      // kick off next chunk's copy so it overlaps this chunk's WMMAs
      if (chunk < (D_ + H_) / KC - 1)
        stage_async(chunk + 1, abuf[(chunk + 1) & 1]);

      const bf16* ab = abuf[chunk & 1];
      const int   kg = chunk * KC;
      const bf16* pW = (kg < D_) ? (pWihRow + kg) : (pWhhRow + (kg - D_));

#pragma unroll
      for (int ks = 0; ks < KC / 32; ++ks) {
        const int kl = ks * 32;
        v16bf a0 = lds_a(ab, m0,     ln, hi, kl);
        v16bf a1 = lds_a(ab, m0 + 1, ln, hi, kl);
        v8bf blo = *(const v8bf*)(pW + kl);
        v8bf bhh = *(const v8bf*)(pW + kl + 16);
        v16bf bm = bf_cat(blo, bhh);
        acc0 = __builtin_amdgcn_wmma_f32_16x16x32_bf16(
                   false, a0, false, bm, (short)0, acc0, false, false);
        acc1 = __builtin_amdgcn_wmma_f32_16x16x32_bf16(
                   false, a1, false, bm, (short)0, acc1, false, false);
      }

      // wait for the overlapped copy, then hand the buffer to all waves
      asm volatile("s_wait_asynccnt 0x0" ::: "memory");
      __syncthreads();
    }

    // ---- scatter gate pre-activations to LDS (C layout: VGPR r -> M=r+8*hi)
#pragma unroll
    for (int r = 0; r < 8; ++r) {
      int b0 = m0 * 16 + r + hi * 8;
      int b1 = (m0 + 1) * 16 + r + hi * 8;
      gbuf[gate * (B_ * 16) + b0 * 16 + ln] = acc0[r];
      gbuf[gate * (B_ * 16) + b1 * 16 + ln] = acc1[r];
    }
    __syncthreads();

    // ---- fused elementwise LSTM cell update (WG-local) ----
    bf16* hwr = hbuf + (size_t)((t + 1) & 1) * (B_ * H_);
#pragma unroll
    for (int e = 0; e < 4; ++e) {
      int idx = tid + e * TPB;
      int b = idx >> 4, jj = idx & 15;
      float iv = sigm (gbuf[0 * (B_*16) + b * 16 + jj]);
      float fv = sigm (gbuf[1 * (B_*16) + b * 16 + jj]);
      float gv = tanhf(gbuf[2 * (B_*16) + b * 16 + jj]);
      float ov = sigm (gbuf[3 * (B_*16) + b * 16 + jj]);
      float c  = fv * creg[e] + iv * gv;
      float h  = ov * tanhf(c);
      creg[e] = c;
      int j = jb + jj;
      hwr[b * H_ + j] = (bf16)h;
      out[(size_t)b * S_ * H_ + (size_t)t * H_ + j] = h;
      if (t == S_ - 1) {
        out[OUTS + (size_t)b * H_ + j] = h;                    // hn
        out[OUTS + (size_t)B_ * H_ + (size_t)b * H_ + j] = c;  // cn
      }
    }

    grid_barrier(cnt, (unsigned)(NWG * (t + 1)));
  }
}

extern "C" void kernel_launch(void* const* d_in, const int* in_sizes, int n_in,
                              void* d_out, int out_size, void* d_ws, size_t ws_size,
                              hipStream_t stream) {
  (void)in_sizes; (void)n_in; (void)out_size; (void)ws_size;
  const float* x   = (const float*)d_in[0];
  const float* h0  = (const float*)d_in[1];
  const float* c0  = (const float*)d_in[2];
  const float* wih = (const float*)d_in[3];
  const float* whh = (const float*)d_in[4];
  const float* bih = (const float*)d_in[5];
  const float* bhh = (const float*)d_in[6];
  float* out = (float*)d_out;

  // workspace layout (~81 MB total)
  char* w = (char*)d_ws;
  bf16*     xbf    = (bf16*)(w);                                   // 64 MiB
  bf16*     Wih_bf = (bf16*)(w + ((size_t)64 << 20));              //  8 MiB
  bf16*     Whh_bf = (bf16*)(w + ((size_t)72 << 20));              //  8 MiB
  bf16*     hbuf   = (bf16*)(w + ((size_t)80 << 20));              // 256 KiB ping-pong
  float*    bsum   = (float*)(w + ((size_t)80 << 20) + ((size_t)512 << 10));
  unsigned* cnt    = (unsigned*)(w + ((size_t)80 << 20) + ((size_t)768 << 10));

  lstm_init<<<(B_ * S_ * D_) / 256, 256, 0, stream>>>(
      x, wih, whh, h0, bih, bhh, xbf, Wih_bf, Whh_bf, hbuf, bsum, cnt);
  lstm_persistent<<<NWG, TPB, 0, stream>>>(
      xbf, c0, Wih_bf, Whh_bf, hbuf, bsum, cnt, out);
}
